// GraphUNet_15839839388405
// MI455X (gfx1250) — compile-verified, matrix operand order checked by reading, not run on hardware
//
#include <hip/hip_runtime.h>
#include <hip/hip_bf16.h>

typedef __attribute__((ext_vector_type(16))) _Float16 v16h;
typedef __attribute__((ext_vector_type(8)))  float    v8f;

#define ENC_NEG_INF 0x007FFFFFu   // order-preserving encoding of -inf

// ---------------------------------------------------------------------------
// helpers
// ---------------------------------------------------------------------------
__device__ __forceinline__ unsigned encf(float f) {
  unsigned u = __float_as_uint(f);
  return (u & 0x80000000u) ? ~u : (u | 0x80000000u);
}
__device__ __forceinline__ float decf(unsigned k) {
  unsigned u = (k & 0x80000000u) ? (k & 0x7FFFFFFFu) : ~k;
  return __uint_as_float(u);
}

// K index inside a 16x32 f16 A/B fragment for element e (0..15), half h (0/1)
__device__ __forceinline__ int kmap16(int e, int h) {
  int v = e >> 1, j = e & 1;
  return (v < 4) ? (8 * h + 2 * v + j) : (16 + 8 * h + 2 * (v - 4) + j);
}

// Load an A fragment (16x32 f16) from row-major LDS buffer, row stride ld.
__device__ __forceinline__ v16h loadA(const _Float16* buf, int ld, int mr, int h, int k0) {
  v16h a;
#pragma unroll
  for (int e = 0; e < 16; ++e) a[e] = buf[mr * ld + k0 + kmap16(e, h)];
  return a;
}

// ---------------------------------------------------------------------------
// weight repack: fp32 [K,N] row-major -> fp16 B-fragments, padded to KPxNP.
// Fragment layout: [(kt*NP/16 + nt)*32 + lane] -> v16h (one lane's 16 halves).
// ---------------------------------------------------------------------------
__global__ void prep_weights(const float* __restrict__ W, _Float16* __restrict__ out,
                             int K, int N, int KP, int NP) {
  int ntiles = NP / 16;
  int total = (KP / 32) * ntiles * 512;
  for (int idx = blockIdx.x * blockDim.x + threadIdx.x; idx < total;
       idx += gridDim.x * blockDim.x) {
    int e    = idx & 15;
    int lane = (idx >> 4) & 31;
    int frag = idx >> 9;
    int nt = frag % ntiles;
    int kt = frag / ntiles;
    int h = lane >> 4;
    int n = nt * 16 + (lane & 15);
    int k = kt * 32 + kmap16(e, h);
    float v = (k < K && n < N) ? W[k * N + n] : 0.0f;
    out[idx] = (_Float16)v;
  }
}

// ---------------------------------------------------------------------------
// fused EdgeConv: feature build -> 3-layer MLP (WMMA) -> segment-max (atomic)
// NI = per-node feature count, KP = padded input K (>= 2*NI, mult of 32)
// ---------------------------------------------------------------------------
template <int NI, int KP>
__global__ __launch_bounds__(256) void edgeconv_wmma(
    const float* __restrict__ nodeF, const int* __restrict__ esrc,
    const int* __restrict__ edst, int E,
    const v16h* __restrict__ W1f, const float* __restrict__ b1,
    const v16h* __restrict__ W2f, const float* __restrict__ b2,
    const v16h* __restrict__ W3f, const float* __restrict__ b3,
    unsigned* __restrict__ outEnc) {
  __shared__ _Float16 sFeat[128 * KP];
  __shared__ _Float16 sHid[128 * 128];
  __shared__ int sDst[128];

  const int t = threadIdx.x;
  const int lane = t & 31;
  const int wave = t >> 5;
  const int e0 = blockIdx.x * 128;

  // ---- build per-edge features: [x_i, x_j - x_i], zero-padded to KP ----
  for (int r = t; r < 128; r += 256) {
    int e = e0 + r;
    int d = -1;
    float fi[NI], fj[NI];
#pragma unroll
    for (int c = 0; c < NI; ++c) { fi[c] = 0.f; fj[c] = 0.f; }
    if (e < E) {
      int s = esrc[e];
      d = edst[e];
#pragma unroll
      for (int c = 0; c < NI; ++c) {
        fi[c] = nodeF[(long)d * NI + c];
        fj[c] = nodeF[(long)s * NI + c];
      }
    }
    sDst[r] = d;
#pragma unroll
    for (int c = 0; c < NI; ++c) {
      sFeat[r * KP + c]      = (_Float16)fi[c];
      sFeat[r * KP + NI + c] = (_Float16)(fj[c] - fi[c]);
    }
#pragma unroll
    for (int c = 2 * NI; c < KP; ++c) sFeat[r * KP + c] = (_Float16)0.f;
  }
  __syncthreads();

  const int m0 = wave * 16;
  const int h  = lane >> 4;
  const int mr = m0 + (lane & 15);
  const int nc = lane & 15;

  v8f acc[8];

  // ---- layer 1: [128,KP] x [KP,128] ----
#pragma unroll
  for (int nt = 0; nt < 8; ++nt) {
    v8f c = {};
#pragma unroll
    for (int kt = 0; kt < KP / 32; ++kt) {
      v16h a = loadA(sFeat, KP, mr, h, kt * 32);
      v16h b = W1f[(kt * 8 + nt) * 32 + lane];
      c = __builtin_amdgcn_wmma_f32_16x16x32_f16(false, a, false, b, (short)0, c,
                                                 false, false);
    }
    acc[nt] = c;
  }
#pragma unroll
  for (int nt = 0; nt < 8; ++nt) {
    int n = nt * 16 + nc;
    float bias = b1[n];
#pragma unroll
    for (int r = 0; r < 8; ++r) {
      int m = m0 + r + 8 * h;
      sHid[m * 128 + n] = (_Float16)fmaxf(acc[nt][r] + bias, 0.f);
    }
  }
  __syncthreads();

  // ---- layer 2: [128,128] x [128,128] ----
#pragma unroll
  for (int nt = 0; nt < 8; ++nt) {
    v8f c = {};
#pragma unroll
    for (int kt = 0; kt < 4; ++kt) {
      v16h a = loadA(sHid, 128, mr, h, kt * 32);
      v16h b = W2f[(kt * 8 + nt) * 32 + lane];
      c = __builtin_amdgcn_wmma_f32_16x16x32_f16(false, a, false, b, (short)0, c,
                                                 false, false);
    }
    acc[nt] = c;
  }
  __syncthreads();
#pragma unroll
  for (int nt = 0; nt < 8; ++nt) {
    int n = nt * 16 + nc;
    float bias = b2[n];
#pragma unroll
    for (int r = 0; r < 8; ++r) {
      int m = m0 + r + 8 * h;
      sHid[m * 128 + n] = (_Float16)fmaxf(acc[nt][r] + bias, 0.f);
    }
  }
  __syncthreads();

  // ---- layer 3: [128,128] x [128,16] (8 real channels) + segment-max ----
  v8f c3 = {};
#pragma unroll
  for (int kt = 0; kt < 4; ++kt) {
    v16h a = loadA(sHid, 128, mr, h, kt * 32);
    v16h b = W3f[kt * 32 + lane];
    c3 = __builtin_amdgcn_wmma_f32_16x16x32_f16(false, a, false, b, (short)0, c3,
                                                false, false);
  }
  if (nc < 8) {
    float bias = b3[nc];
#pragma unroll
    for (int r = 0; r < 8; ++r) {
      int m = m0 + r + 8 * h;
      int d = sDst[m];
      if (d >= 0) atomicMax(&outEnc[(long)d * 8 + nc], encf(c3[r] + bias));
    }
  }
}

// ---------------------------------------------------------------------------
// final per-node MLP: [N,8] -> 128 -> 128 -> 1 (relu between), fp32 out
// ---------------------------------------------------------------------------
__global__ __launch_bounds__(256) void node_mlp_wmma(
    const float* __restrict__ nodeF, int N,
    const v16h* __restrict__ W1f, const float* __restrict__ b1,
    const v16h* __restrict__ W2f, const float* __restrict__ b2,
    const v16h* __restrict__ W3f, const float* __restrict__ b3,
    float* __restrict__ out) {
  __shared__ _Float16 sFeat[128 * 32];
  __shared__ _Float16 sHid[128 * 128];

  const int t = threadIdx.x;
  const int lane = t & 31;
  const int wave = t >> 5;
  const int base = blockIdx.x * 128;

  for (int r = t; r < 128; r += 256) {
    int i = base + r;
#pragma unroll
    for (int c = 0; c < 8; ++c)
      sFeat[r * 32 + c] = (_Float16)((i < N) ? nodeF[(long)i * 8 + c] : 0.f);
#pragma unroll
    for (int c = 8; c < 32; ++c) sFeat[r * 32 + c] = (_Float16)0.f;
  }
  __syncthreads();

  const int m0 = wave * 16;
  const int h  = lane >> 4;
  const int mr = m0 + (lane & 15);
  const int nc = lane & 15;

  v8f acc[8];
#pragma unroll
  for (int nt = 0; nt < 8; ++nt) {
    v16h a = loadA(sFeat, 32, mr, h, 0);
    v16h b = W1f[nt * 32 + lane];
    v8f c = {};
    c = __builtin_amdgcn_wmma_f32_16x16x32_f16(false, a, false, b, (short)0, c,
                                               false, false);
    acc[nt] = c;
  }
#pragma unroll
  for (int nt = 0; nt < 8; ++nt) {
    int n = nt * 16 + nc;
    float bias = b1[n];
#pragma unroll
    for (int r = 0; r < 8; ++r) {
      int m = m0 + r + 8 * h;
      sHid[m * 128 + n] = (_Float16)fmaxf(acc[nt][r] + bias, 0.f);
    }
  }
  __syncthreads();

#pragma unroll
  for (int nt = 0; nt < 8; ++nt) {
    v8f c = {};
#pragma unroll
    for (int kt = 0; kt < 4; ++kt) {
      v16h a = loadA(sHid, 128, mr, h, kt * 32);
      v16h b = W2f[(kt * 8 + nt) * 32 + lane];
      c = __builtin_amdgcn_wmma_f32_16x16x32_f16(false, a, false, b, (short)0, c,
                                                 false, false);
    }
    acc[nt] = c;
  }
  __syncthreads();
#pragma unroll
  for (int nt = 0; nt < 8; ++nt) {
    int n = nt * 16 + nc;
    float bias = b2[n];
#pragma unroll
    for (int r = 0; r < 8; ++r) {
      int m = m0 + r + 8 * h;
      sHid[m * 128 + n] = (_Float16)fmaxf(acc[nt][r] + bias, 0.f);
    }
  }
  __syncthreads();

  v8f c3 = {};
#pragma unroll
  for (int kt = 0; kt < 4; ++kt) {
    v16h a = loadA(sHid, 128, mr, h, kt * 32);
    v16h b = W3f[kt * 32 + lane];
    c3 = __builtin_amdgcn_wmma_f32_16x16x32_f16(false, a, false, b, (short)0, c3,
                                                false, false);
  }
  if (nc == 0) {
    float bias = b3[0];
#pragma unroll
    for (int r = 0; r < 8; ++r) {
      int i = base + m0 + r + 8 * h;
      if (i < N) out[i] = c3[r] + bias;
    }
  }
}

// ---------------------------------------------------------------------------
// elementwise / reduction helpers
// ---------------------------------------------------------------------------
__global__ void fill_u32(unsigned* p, unsigned v, int n) {
  int i = blockIdx.x * blockDim.x + threadIdx.x;
  if (i < n) p[i] = v;
}
__global__ void zero_f32(float* p, int n) {
  int i = blockIdx.x * blockDim.x + threadIdx.x;
  if (i < n) p[i] = 0.f;
}

// decode seg-max (fill -inf -> 0), relu, accumulate per-channel sum/sumsq
__global__ void decode_relu_stats(const unsigned* __restrict__ enc,
                                  float* __restrict__ out,
                                  float* __restrict__ stats, int total) {
  __shared__ float ssum[8], ssq[8];
  if (threadIdx.x < 8) { ssum[threadIdx.x] = 0.f; ssq[threadIdx.x] = 0.f; }
  __syncthreads();
  int i = blockIdx.x * blockDim.x + threadIdx.x;
  if (i < total) {
    unsigned k = enc[i];
    float f = (k == ENC_NEG_INF) ? 0.f : decf(k);
    f = fmaxf(f, 0.f);
    out[i] = f;
    atomicAdd(&ssum[i & 7], f);
    atomicAdd(&ssq[i & 7], f * f);
  }
  __syncthreads();
  if (threadIdx.x < 8) {
    atomicAdd(&stats[threadIdx.x], ssum[threadIdx.x]);
    atomicAdd(&stats[8 + threadIdx.x], ssq[threadIdx.x]);
  }
}

__global__ void bn_finalize(const float* __restrict__ stats,
                            const float* __restrict__ gamma,
                            const float* __restrict__ beta,
                            float* __restrict__ sc, float n) {
  int c = threadIdx.x;
  if (c < 8) {
    float mu = stats[c] / n;
    float var = stats[8 + c] / n - mu * mu;
    float s = gamma[c] * rsqrtf(var + 1e-5f);
    sc[c] = s;
    sc[8 + c] = beta[c] - mu * s;
  }
}

__global__ void bn_apply(float* __restrict__ buf, const float* __restrict__ sc,
                         int total) {
  int i = blockIdx.x * blockDim.x + threadIdx.x;
  if (i < total) {
    int c = i & 7;
    buf[i] = buf[i] * sc[c] + sc[8 + c];
  }
}

__global__ void pool_accum(const float* __restrict__ h, const int* __restrict__ cl,
                           float* __restrict__ sum, float* __restrict__ cnt, int N) {
  int i = blockIdx.x * blockDim.x + threadIdx.x;
  if (i < N) {
    int c = cl[i];
#pragma unroll
    for (int k = 0; k < 8; ++k) atomicAdd(&sum[(long)c * 8 + k], h[(long)i * 8 + k]);
    atomicAdd(&cnt[c], 1.0f);
  }
}

__global__ void pool_div(const float* __restrict__ sum, const float* __restrict__ cnt,
                         float* __restrict__ xp, int total) {
  int i = blockIdx.x * blockDim.x + threadIdx.x;
  if (i < total) xp[i] = sum[i] / fmaxf(cnt[i >> 3], 1.0f);
}

__global__ void assemble_nodeF1(const float* __restrict__ x1,
                                const float* __restrict__ x1p,
                                float* __restrict__ out, int N1) {
  int i = blockIdx.x * blockDim.x + threadIdx.x;
  if (i < N1) {
    out[(long)i * 10 + 0] = x1[(long)i * 2 + 0];
    out[(long)i * 10 + 1] = x1[(long)i * 2 + 1];
#pragma unroll
    for (int c = 0; c < 8; ++c) out[(long)i * 10 + 2 + c] = x1p[(long)i * 8 + c];
  }
}

__global__ void assemble_nodeFu(const float* __restrict__ x1,
                                const float* __restrict__ x1p,
                                const float* __restrict__ X1,
                                float* __restrict__ out, int N1) {
  int i = blockIdx.x * blockDim.x + threadIdx.x;
  if (i < N1) {
    out[(long)i * 18 + 0] = x1[(long)i * 2 + 0];
    out[(long)i * 18 + 1] = x1[(long)i * 2 + 1];
#pragma unroll
    for (int c = 0; c < 8; ++c) {
      out[(long)i * 18 + 2 + c]  = x1p[(long)i * 8 + c];
      out[(long)i * 18 + 10 + c] = X1[(long)i * 8 + c];
    }
  }
}

__global__ void assemble_nodeF0(const float* __restrict__ x0,
                                const float* __restrict__ h0,
                                const float* __restrict__ Xu,
                                const int* __restrict__ cl,
                                float* __restrict__ out, int N0) {
  int i = blockIdx.x * blockDim.x + threadIdx.x;
  if (i < N0) {
    int c0 = cl[i];
    out[(long)i * 18 + 0] = x0[(long)i * 3 + 0];
    out[(long)i * 18 + 1] = x0[(long)i * 3 + 1];
#pragma unroll
    for (int c = 0; c < 8; ++c) {
      out[(long)i * 18 + 2 + c]  = h0[(long)i * 8 + c];
      out[(long)i * 18 + 10 + c] = Xu[(long)c0 * 8 + c];
    }
  }
}

// ---------------------------------------------------------------------------
// host launcher
// ---------------------------------------------------------------------------
extern "C" void kernel_launch(void* const* d_in, const int* in_sizes, int n_in,
                              void* d_out, int out_size, void* d_ws, size_t ws_size,
                              hipStream_t stream) {
  (void)in_sizes; (void)n_in; (void)out_size; (void)ws_size;
  const int N0 = 100000, E0 = 1000000, N1 = 20000, E1 = 200000;

  const float* x0 = (const float*)d_in[0];
  const float* x1 = (const float*)d_in[1];
  const int* ei0 = (const int*)d_in[2];
  const int* ei1 = (const int*)d_in[3];
  const int* cluster0 = (const int*)d_in[4];
  // params flattened alphabetically: Lconv0, Lconv1, Lnorm0, Lnorm1,
  // Rconv0, Rconv1, Rnorm0, Rnorm1, mlp_out; each conv = (W,b) x3
  const float* Lc0W[3] = {(const float*)d_in[5], (const float*)d_in[7], (const float*)d_in[9]};
  const float* Lc0b[3] = {(const float*)d_in[6], (const float*)d_in[8], (const float*)d_in[10]};
  const float* Lc1W[3] = {(const float*)d_in[11], (const float*)d_in[13], (const float*)d_in[15]};
  const float* Lc1b[3] = {(const float*)d_in[12], (const float*)d_in[14], (const float*)d_in[16]};
  const float* Ln0g = (const float*)d_in[17]; const float* Ln0b = (const float*)d_in[18];
  const float* Ln1g = (const float*)d_in[19]; const float* Ln1b = (const float*)d_in[20];
  const float* Rc0W[3] = {(const float*)d_in[21], (const float*)d_in[23], (const float*)d_in[25]};
  const float* Rc0b[3] = {(const float*)d_in[22], (const float*)d_in[24], (const float*)d_in[26]};
  const float* Rc1W[3] = {(const float*)d_in[27], (const float*)d_in[29], (const float*)d_in[31]};
  const float* Rc1b[3] = {(const float*)d_in[28], (const float*)d_in[30], (const float*)d_in[32]};
  const float* Rn0g = (const float*)d_in[33]; const float* Rn0b = (const float*)d_in[34];
  const float* Rn1g = (const float*)d_in[35]; const float* Rn1b = (const float*)d_in[36];
  const float* MoW[3] = {(const float*)d_in[37], (const float*)d_in[39], (const float*)d_in[41]};
  const float* Mob[3] = {(const float*)d_in[38], (const float*)d_in[40], (const float*)d_in[42]};

  const int* src0 = ei0;        const int* dst0 = ei0 + E0;
  const int* src1 = ei1;        const int* dst1 = ei1 + E1;

  // ---- workspace carving ----
  char* ws = (char*)d_ws;
  size_t off = 0;
  auto carve = [&](size_t bytes) -> void* {
    void* p = ws + off;
    off += (bytes + 255) & ~(size_t)255;
    return p;
  };
  auto fragBytes = [](int KP, int NP) -> size_t {
    return (size_t)(KP / 32) * (NP / 16) * 512 * sizeof(_Float16);
  };

  _Float16* wL0[3] = {(_Float16*)carve(fragBytes(32, 128)),
                      (_Float16*)carve(fragBytes(128, 128)),
                      (_Float16*)carve(fragBytes(128, 16))};
  _Float16* wL1[3] = {(_Float16*)carve(fragBytes(32, 128)),
                      (_Float16*)carve(fragBytes(128, 128)),
                      (_Float16*)carve(fragBytes(128, 16))};
  _Float16* wR0[3] = {(_Float16*)carve(fragBytes(64, 128)),
                      (_Float16*)carve(fragBytes(128, 128)),
                      (_Float16*)carve(fragBytes(128, 16))};
  _Float16* wR1[3] = {(_Float16*)carve(fragBytes(64, 128)),
                      (_Float16*)carve(fragBytes(128, 128)),
                      (_Float16*)carve(fragBytes(128, 16))};
  _Float16* wMo[3] = {(_Float16*)carve(fragBytes(32, 128)),
                      (_Float16*)carve(fragBytes(128, 128)),
                      (_Float16*)carve(fragBytes(128, 16))};

  unsigned* enc0 = (unsigned*)carve((size_t)N0 * 8 * 4);
  unsigned* enc1 = (unsigned*)carve((size_t)N1 * 8 * 4);
  float* h0bn   = (float*)carve((size_t)N0 * 8 * 4);
  float* stats  = (float*)carve(16 * 4);
  float* sc     = (float*)carve(16 * 4);
  float* psum   = (float*)carve((size_t)N1 * 8 * 4);
  float* pcnt   = (float*)carve((size_t)N1 * 4);
  float* x1p    = (float*)carve((size_t)N1 * 8 * 4);
  float* nodeF1 = (float*)carve((size_t)N1 * 10 * 4);
  float* X1bn   = (float*)carve((size_t)N1 * 8 * 4);
  float* nodeFu = (float*)carve((size_t)N1 * 18 * 4);
  float* Xubn   = (float*)carve((size_t)N1 * 8 * 4);
  float* nodeF0 = (float*)carve((size_t)N0 * 18 * 4);
  float* Ybn    = (float*)carve((size_t)N0 * 8 * 4);

  auto g1 = [](int n) { return dim3((unsigned)((n + 255) / 256)); };
  const dim3 B(256);

  // ---- weight repack (fp32 -> fp16 WMMA B-fragments) ----
  auto prep = [&](const float* W, _Float16* dstp, int K, int N, int KP, int NP) {
    int total = (KP / 32) * (NP / 16) * 512;
    prep_weights<<<g1(total), B, 0, stream>>>(W, dstp, K, N, KP, NP);
  };
  prep(Lc0W[0], wL0[0], 6, 128, 32, 128);  prep(Lc0W[1], wL0[1], 128, 128, 128, 128);  prep(Lc0W[2], wL0[2], 128, 8, 128, 16);
  prep(Lc1W[0], wL1[0], 20, 128, 32, 128); prep(Lc1W[1], wL1[1], 128, 128, 128, 128);  prep(Lc1W[2], wL1[2], 128, 8, 128, 16);
  prep(Rc0W[0], wR0[0], 36, 128, 64, 128); prep(Rc0W[1], wR0[1], 128, 128, 128, 128);  prep(Rc0W[2], wR0[2], 128, 8, 128, 16);
  prep(Rc1W[0], wR1[0], 36, 128, 64, 128); prep(Rc1W[1], wR1[1], 128, 128, 128, 128);  prep(Rc1W[2], wR1[2], 128, 8, 128, 16);
  prep(MoW[0],  wMo[0], 8, 128, 32, 128);  prep(MoW[1],  wMo[1], 128, 128, 128, 128);  prep(MoW[2],  wMo[2], 128, 1, 128, 16);

  // ---- level 0 down: EdgeConv(Lconv0) over E0, relu+BN, avg-pool ----
  fill_u32<<<g1(N0 * 8), B, 0, stream>>>(enc0, ENC_NEG_INF, N0 * 8);
  edgeconv_wmma<3, 32><<<g1(E0 * 2), B, 0, stream>>>(  // (E0+127)/128 blocks == g1(E0*2)
      x0, src0, dst0, E0,
      (const v16h*)wL0[0], Lc0b[0], (const v16h*)wL0[1], Lc0b[1],
      (const v16h*)wL0[2], Lc0b[2], enc0);
  zero_f32<<<dim3(1), B, 0, stream>>>(stats, 16);
  decode_relu_stats<<<g1(N0 * 8), B, 0, stream>>>(enc0, h0bn, stats, N0 * 8);
  bn_finalize<<<dim3(1), dim3(8), 0, stream>>>(stats, Ln0g, Ln0b, sc, (float)N0);
  bn_apply<<<g1(N0 * 8), B, 0, stream>>>(h0bn, sc, N0 * 8);

  zero_f32<<<g1(N1 * 8), B, 0, stream>>>(psum, N1 * 8);
  zero_f32<<<g1(N1), B, 0, stream>>>(pcnt, N1);
  pool_accum<<<g1(N0), B, 0, stream>>>(h0bn, cluster0, psum, pcnt, N0);
  pool_div<<<g1(N1 * 8), B, 0, stream>>>(psum, pcnt, x1p, N1 * 8);

  // ---- bottom level: EdgeConv(Lconv1) over E1 ----
  assemble_nodeF1<<<g1(N1), B, 0, stream>>>(x1, x1p, nodeF1, N1);
  fill_u32<<<g1(N1 * 8), B, 0, stream>>>(enc1, ENC_NEG_INF, N1 * 8);
  edgeconv_wmma<10, 32><<<g1(E1 * 2), B, 0, stream>>>(
      nodeF1, src1, dst1, E1,
      (const v16h*)wL1[0], Lc1b[0], (const v16h*)wL1[1], Lc1b[1],
      (const v16h*)wL1[2], Lc1b[2], enc1);
  zero_f32<<<dim3(1), B, 0, stream>>>(stats, 16);
  decode_relu_stats<<<g1(N1 * 8), B, 0, stream>>>(enc1, X1bn, stats, N1 * 8);
  bn_finalize<<<dim3(1), dim3(8), 0, stream>>>(stats, Ln1g, Ln1b, sc, (float)N1);
  bn_apply<<<g1(N1 * 8), B, 0, stream>>>(X1bn, sc, N1 * 8);

  // ---- up path j=1: EdgeConv(Rconv1) over E1 ----
  assemble_nodeFu<<<g1(N1), B, 0, stream>>>(x1, x1p, X1bn, nodeFu, N1);
  fill_u32<<<g1(N1 * 8), B, 0, stream>>>(enc1, ENC_NEG_INF, N1 * 8);
  edgeconv_wmma<18, 64><<<g1(E1 * 2), B, 0, stream>>>(
      nodeFu, src1, dst1, E1,
      (const v16h*)wR1[0], Rc1b[0], (const v16h*)wR1[1], Rc1b[1],
      (const v16h*)wR1[2], Rc1b[2], enc1);
  zero_f32<<<dim3(1), B, 0, stream>>>(stats, 16);
  decode_relu_stats<<<g1(N1 * 8), B, 0, stream>>>(enc1, Xubn, stats, N1 * 8);
  bn_finalize<<<dim3(1), dim3(8), 0, stream>>>(stats, Rn1g, Rn1b, sc, (float)N1);
  bn_apply<<<g1(N1 * 8), B, 0, stream>>>(Xubn, sc, N1 * 8);

  // ---- up path j=0: EdgeConv(Rconv0) over E0 (with unpool gather) ----
  assemble_nodeF0<<<g1(N0), B, 0, stream>>>(x0, h0bn, Xubn, cluster0, nodeF0, N0);
  fill_u32<<<g1(N0 * 8), B, 0, stream>>>(enc0, ENC_NEG_INF, N0 * 8);
  edgeconv_wmma<18, 64><<<g1(E0 * 2), B, 0, stream>>>(
      nodeF0, src0, dst0, E0,
      (const v16h*)wR0[0], Rc0b[0], (const v16h*)wR0[1], Rc0b[1],
      (const v16h*)wR0[2], Rc0b[2], enc0);
  zero_f32<<<dim3(1), B, 0, stream>>>(stats, 16);
  decode_relu_stats<<<g1(N0 * 8), B, 0, stream>>>(enc0, Ybn, stats, N0 * 8);
  bn_finalize<<<dim3(1), dim3(8), 0, stream>>>(stats, Rn0g, Rn0b, sc, (float)N0);
  bn_apply<<<g1(N0 * 8), B, 0, stream>>>(Ybn, sc, N0 * 8);

  // ---- final per-node MLP: [N0,8] -> 128 -> 128 -> 1 ----
  node_mlp_wmma<<<dim3((unsigned)((N0 + 127) / 128)), B, 0, stream>>>(
      Ybn, N0,
      (const v16h*)wMo[0], Mob[0], (const v16h*)wMo[1], Mob[1],
      (const v16h*)wMo[2], Mob[2], (float*)d_out);
}
